// STPAttention_57732950393319
// MI455X (gfx1250) — matrix operand order repaired and checked by reading, not compiled, not run on hardware
//
#include <hip/hip_runtime.h>
#include <hip/hip_bf16.h>
#include <math.h>

// ---------------- types ----------------
typedef __attribute__((ext_vector_type(16))) __bf16 v16bf;
typedef __attribute__((ext_vector_type(8)))  __bf16 v8bf;
typedef __attribute__((ext_vector_type(8)))  float  v8f;
typedef __attribute__((ext_vector_type(4)))  unsigned int uint4v;
typedef __attribute__((ext_vector_type(2)))  unsigned int uint2v;
typedef __attribute__((ext_vector_type(4)))  int int4v;
typedef unsigned short ushort_t;

#define GLOBAL_AS __attribute__((address_space(1)))
#define LDS_AS    __attribute__((address_space(3)))

#define B_  4
#define L_  4096
#define D_  1024
#define H_  16
#define d_  64

#if defined(__has_builtin)
#if __has_builtin(__builtin_amdgcn_global_load_async_to_lds_b128) && \
    __has_builtin(__builtin_amdgcn_s_wait_asynccnt)
#define USE_ASYNC_LDS 1
#endif
#endif

// ---------------- f32 -> bf16 (RNE) ----------------
__device__ __forceinline__ ushort_t f2bf(float f) {
    unsigned u = __float_as_uint(f);
    unsigned r = (u + 0x7FFFu + ((u >> 16) & 1u)) >> 16;
    return (ushort_t)r;
}

__global__ void cvt_f32_bf16(const float* __restrict__ s, ushort_t* __restrict__ dst, int n) {
    for (int i = blockIdx.x * blockDim.x + threadIdx.x; i < n; i += gridDim.x * blockDim.x)
        dst[i] = f2bf(s[i]);
}

// ---------------- bf16 WMMA GEMM: C[M,N] = A[M,K] * Bt[N,K]^T + bias[N] ----------------
// A, Bt bf16 K-major (lda = ldb = K). C f32 row-major. M%128==0, N%128==0, K%64==0.
// Block: 256 threads (8 waves). Block tile 128x128, K-step 64.
// Wave tile 32x64 -> 2x4 accumulators, 16 wmma per K-step.
// Double-buffered LDS fed by GLOBAL_LOAD_ASYNC_TO_LDS_B128 (ASYNCcnt) when available.
#define LDT 72   // LDS row stride in halves (64 data + 8 pad) -> 144B rows, 16B aligned

__device__ __forceinline__ void cp16_async(const ushort_t* g, ushort_t* l) {
#if defined(USE_ASYNC_LDS)
    __builtin_amdgcn_global_load_async_to_lds_b128(
        (GLOBAL_AS int4v*)g, (LDS_AS int4v*)l, 0, 0);
#else
    *(uint4v*)l = *(const uint4v*)g;
#endif
}

__global__ __launch_bounds__(256)
void gemm_bf16(const ushort_t* __restrict__ A, const ushort_t* __restrict__ Bt,
               const float* __restrict__ bias, float* __restrict__ C,
               int M, int N, int K) {
    __shared__ ushort_t lA[2][128 * LDT];
    __shared__ ushort_t lB[2][128 * LDT];

    const int t    = threadIdx.x;
    const int lane = t & 31;
    const int w    = t >> 5;            // 0..7
    const int m0   = (w >> 1) * 32;     // wave M origin in tile
    const int n0   = (w & 1) * 64;      // wave N origin in tile
    const int rowBase = blockIdx.y * 128;
    const int colBase = blockIdx.x * 128;

    v8f acc[2][4] = {};

    const int lr  = lane & 15;          // fragment row within 16
    const int khA = (lane >> 4) * 16;   // A frag K base within 32-slice (halves)
    const int khB = (lane >> 4) * 16;   // B frag K base within 32-slice (halves)

    // per-thread staging chunks: 128 rows x 64 halves / 16B = 1024 chunks per matrix,
    // 4 chunks per thread per matrix: row = c>>3, seg = (c&7)*8 halves
    const int nT = K >> 6;

#if defined(USE_ASYNC_LDS)
    // prologue: issue tile 0 into buffer 0
    #pragma unroll
    for (int r = 0; r < 4; ++r) {
        int c = t + (r << 8), row = c >> 3, seg = (c & 7) * 8;
        cp16_async(A  + (size_t)(rowBase + row) * K + seg, &lA[0][row * LDT + seg]);
        cp16_async(Bt + (size_t)(colBase + row) * K + seg, &lB[0][row * LDT + seg]);
    }
#endif

    int p = 0;
    for (int kt = 0; kt < nT; ++kt, p ^= 1) {
#if defined(USE_ASYNC_LDS)
        if (kt + 1 < nT) {               // issue next tile into other buffer
            int k0n = (kt + 1) << 6;
            #pragma unroll
            for (int r = 0; r < 4; ++r) {
                int c = t + (r << 8), row = c >> 3, seg = (c & 7) * 8;
                cp16_async(A  + (size_t)(rowBase + row) * K + k0n + seg,
                           &lA[p ^ 1][row * LDT + seg]);
                cp16_async(Bt + (size_t)(colBase + row) * K + k0n + seg,
                           &lB[p ^ 1][row * LDT + seg]);
            }
            __builtin_amdgcn_s_wait_asynccnt(8);   // tile kt fully landed
        } else {
            __builtin_amdgcn_s_wait_asynccnt(0);
        }
#else
        {   // synchronous fallback: stage current tile into buffer p
            int k0 = kt << 6;
            #pragma unroll
            for (int r = 0; r < 4; ++r) {
                int c = t + (r << 8), row = c >> 3, seg = (c & 7) * 8;
                cp16_async(A  + (size_t)(rowBase + row) * K + k0 + seg, &lA[p][row * LDT + seg]);
                cp16_async(Bt + (size_t)(colBase + row) * K + k0 + seg, &lB[p][row * LDT + seg]);
            }
        }
#endif
        __syncthreads();

        #pragma unroll
        for (int kk = 0; kk < 64; kk += 32) {
            v16bf afr[2];
            #pragma unroll
            for (int mi = 0; mi < 2; ++mi) {
                const __bf16* pa = (const __bf16*)&lA[p][(m0 + mi * 16 + lr) * LDT + kk + khA];
                v8bf lo = *(const v8bf*)pa;           // K = kh .. kh+7
                v8bf hi = *(const v8bf*)(pa + 8);     // K = kh+8 .. kh+15
                afr[mi] = __builtin_shufflevector(lo, hi,
                    0,1,2,3,4,5,6,7,8,9,10,11,12,13,14,15);
            }
            v16bf bfr[4];
            #pragma unroll
            for (int ni = 0; ni < 4; ++ni) {
                const __bf16* pb = (const __bf16*)&lB[p][(n0 + ni * 16 + lr) * LDT + kk + khB];
                v8bf lo = *(const v8bf*)pb;           // 16 contiguous K values
                v8bf hi = *(const v8bf*)(pb + 8);
                bfr[ni] = __builtin_shufflevector(lo, hi,
                    0,1,2,3,4,5,6,7,8,9,10,11,12,13,14,15);
            }
            #pragma unroll
            for (int mi = 0; mi < 2; ++mi)
                #pragma unroll
                for (int ni = 0; ni < 4; ++ni)
                    acc[mi][ni] = __builtin_amdgcn_wmma_f32_16x16x32_bf16(
                        false, afr[mi], false, bfr[ni], (short)0, acc[mi][ni], false, false);
        }
        __syncthreads();   // reads of buf p done before it is overwritten
    }

    // ---- epilogue: C VGPR r holds M=r (lanes 0-15) / M=r+8 (lanes 16-31), N=lane&15 ----
    const int mAdd = (lane >> 4) * 8;
    #pragma unroll
    for (int mi = 0; mi < 2; ++mi) {
        #pragma unroll
        for (int ni = 0; ni < 4; ++ni) {
            int n = colBase + n0 + ni * 16 + lr;
            float bv = bias[n];
            #pragma unroll
            for (int r = 0; r < 8; ++r) {
                int m = rowBase + m0 + mi * 16 + r + mAdd;
                C[(size_t)m * N + n] = acc[mi][ni][r] + bv;
            }
        }
    }
}

// ---------------- sequential fast-weight scan ----------------
// grid = B*H blocks, 256 threads. Thread t owns state row i = t>>2, cols (t&3)*16 .. +15.
// 16 time steps of q/k/v staged per barrier (float4 loads); j-reduction via wave32
// shuffles; output staged in LDS, flushed as packed bf16 (8B stores).
__global__ __launch_bounds__(256)
void stp_scan(const float* __restrict__ qkv,   // (B,L,3,H,d) f32
              const float* __restrict__ Lam,   // (H,d,d)
              const float* __restrict__ Gam,   // (H,d,d)
              const float* __restrict__ Wst,   // (H,d,d)
              ushort_t* __restrict__ ybf)      // (B,L,H*d) bf16
{
    const int bh = blockIdx.x;
    const int b  = bh >> 4;
    const int h  = bh & 15;
    const int t  = threadIdx.x;
    const int i  = t >> 2;            // state row 0..63
    const int js = (t & 3) * 16;      // state col segment

    float ret[16], gam[16], ws[16], st[16];
    {
        size_t base = (size_t)h * (d_ * d_) + i * d_ + js;
        #pragma unroll
        for (int q4 = 0; q4 < 4; ++q4) {
            float4 lv = *(const float4*)(Lam + base + q4 * 4);
            float4 gv = *(const float4*)(Gam + base + q4 * 4);
            float4 wv = *(const float4*)(Wst + base + q4 * 4);
            ret[q4*4+0] = 1.0f / (1.0f + __expf(lv.x));   // 1 - sigmoid
            ret[q4*4+1] = 1.0f / (1.0f + __expf(lv.y));
            ret[q4*4+2] = 1.0f / (1.0f + __expf(lv.z));
            ret[q4*4+3] = 1.0f / (1.0f + __expf(lv.w));
            gam[q4*4+0] = gv.x; gam[q4*4+1] = gv.y; gam[q4*4+2] = gv.z; gam[q4*4+3] = gv.w;
            ws[q4*4+0]  = wv.x; ws[q4*4+1]  = wv.y; ws[q4*4+2]  = wv.z; ws[q4*4+3]  = wv.w;
        }
        #pragma unroll
        for (int jj = 0; jj < 16; ++jj) st[jj] = 0.0f;
    }

    __shared__ float qs[16][64], ks[16][64], vs[16][64];
    __shared__ float ybuf[16][64];
    const float scale = 0.125f;       // 1/sqrt(64)

    for (int c = 0; c < L_ / 16; ++c) {
        const int t0 = c * 16;
        // stage 16 steps of q,k,v: 3072 floats = 768 float4, 3 per thread, coalesced
        #pragma unroll
        for (int r = 0; r < 3; ++r) {
            int idx = t + (r << 8);            // 0..767
            int off = idx << 2;                // 0..3068
            int stp = off / 192;
            int rem = off - stp * 192;
            int s   = rem >> 6;
            int j   = rem & 63;
            float4 val = *(const float4*)(qkv +
                (size_t)(b * L_ + t0 + stp) * (3 * D_) + s * D_ + h * d_ + j);
            float* dstrow = (s == 0) ? &qs[stp][0] : (s == 1) ? &ks[stp][0] : &vs[stp][0];
            if (s == 1) { val.x *= scale; val.y *= scale; val.z *= scale; val.w *= scale; }
            *(float4*)&dstrow[j] = val;
        }
        __syncthreads();

        for (int stp = 0; stp < 16; ++stp) {
            float vi = vs[stp][i];
            float part = 0.0f;
            #pragma unroll
            for (int jj = 0; jj < 16; ++jj) {
                float kj = ks[stp][js + jj];
                st[jj] = ret[jj] * st[jj] + gam[jj] * (vi * kj);
                part  += (ws[jj] + st[jj]) * qs[stp][js + jj];
            }
            part += __shfl_xor(part, 1, 32);
            part += __shfl_xor(part, 2, 32);
            if ((t & 3) == 0) ybuf[stp][i] = part;
        }
        __syncthreads();

        // flush 16 steps of y as packed bf16: 1024 values = 256 x (4 bf16 = 8B stores)
        {
            int off = t << 2;                  // 0..1020
            int stp = off >> 6;
            int j   = off & 63;
            const float* row = &ybuf[stp][0];
            uint2v pk;
            pk.x = (unsigned)f2bf(row[j])     | ((unsigned)f2bf(row[j + 1]) << 16);
            pk.y = (unsigned)f2bf(row[j + 2]) | ((unsigned)f2bf(row[j + 3]) << 16);
            *(uint2v*)(ybf + (size_t)(b * L_ + t0 + stp) * D_ + h * d_ + j) = pk;
        }
        __syncthreads();
    }
}

// ---------------- launch ----------------
extern "C" void kernel_launch(void* const* d_in, const int* in_sizes, int n_in,
                              void* d_out, int out_size, void* d_ws, size_t ws_size,
                              hipStream_t stream) {
    const float* x       = (const float*)d_in[0];  // (B,L,D)
    const float* Wqkv_w  = (const float*)d_in[1];  // (3D,D)
    const float* Wqkv_b  = (const float*)d_in[2];  // (3D)
    const float* out_w   = (const float*)d_in[3];  // (D,D)
    const float* out_b   = (const float*)d_in[4];  // (D)
    const float* W_st    = (const float*)d_in[5];  // (H,d,d)
    const float* Lam     = (const float*)d_in[6];  // (H,d,d)
    const float* Gam     = (const float*)d_in[7];  // (H,d,d)
    float* out = (float*)d_out;                    // (B,L,D) f32

    const int M  = B_ * L_;        // 16384
    const int N1 = 3 * D_;         // 3072
    const int K  = D_;             // 1024

    // workspace carve-up
    char* ws = (char*)d_ws;
    float*    qkv    = (float*)ws;     ws += (size_t)M * N1 * sizeof(float);      // 192 MB
    ushort_t* x_bf   = (ushort_t*)ws;  ws += (size_t)M * D_ * sizeof(ushort_t);   //  32 MB
    ushort_t* wq_bf  = (ushort_t*)ws;  ws += (size_t)N1 * D_ * sizeof(ushort_t);  //   6 MB
    ushort_t* wo_bf  = (ushort_t*)ws;  ws += (size_t)D_ * D_ * sizeof(ushort_t);  //   2 MB
    ushort_t* y_bf   = (ushort_t*)ws;  ws += (size_t)M * D_ * sizeof(ushort_t);   //  32 MB

    // 1) convert fp32 operands to bf16
    cvt_f32_bf16<<<4096, 256, 0, stream>>>(x,      x_bf,  M * D_);
    cvt_f32_bf16<<<2048, 256, 0, stream>>>(Wqkv_w, wq_bf, N1 * D_);
    cvt_f32_bf16<<<1024, 256, 0, stream>>>(out_w,  wo_bf, D_ * D_);

    // 2) qkv = x @ Wqkv^T + b    (16384 x 3072)
    gemm_bf16<<<dim3(N1 / 128, M / 128), 256, 0, stream>>>(x_bf, wq_bf, Wqkv_b, qkv, M, N1, K);

    // 3) recurrent fast-weight scan -> y (bf16)
    stp_scan<<<B_ * H_, 256, 0, stream>>>(qkv, Lam, Gam, W_st, y_bf);

    // 4) out = y @ out_w^T + b   (16384 x 1024)
    gemm_bf16<<<dim3(D_ / 128, M / 128), 256, 0, stream>>>(y_bf, wo_bf, out_b, out, M, D_, K);
}
